// CSA_23081154248927
// MI455X (gfx1250) — matrix operand order repaired and checked by reading, not compiled, not run on hardware
//
#include <hip/hip_runtime.h>
#include <hip/hip_bf16.h>

typedef __attribute__((ext_vector_type(16))) _Float16 v16h;
typedef __attribute__((ext_vector_type(8)))  _Float16 v8h;
typedef __attribute__((ext_vector_type(8)))  float    v8f;

#define EPS_CSA 1e-8f
#define EPS_BN  1e-5f

struct Frag { v16h a, b0, b1, b2, b3; };

static __device__ __forceinline__ v16h cat16(v8h lo, v8h hi) {
    return __builtin_shufflevector(lo, hi, 0,1,2,3,4,5,6,7,8,9,10,11,12,13,14,15);
}

static __device__ __forceinline__ Frag load_frag(
    const _Float16* __restrict__ Arow,
    const _Float16* __restrict__ B0, const _Float16* __restrict__ B1,
    const _Float16* __restrict__ B2, const _Float16* __restrict__ B3, int k0)
{
    v8h alo = *(const v8h*)(Arow + k0);
    v8h ahi = *(const v8h*)(Arow + k0 + 8);
    v8h p0  = *(const v8h*)(B0 + k0);
    v8h q0  = *(const v8h*)(B0 + k0 + 8);
    v8h p1  = *(const v8h*)(B1 + k0);
    v8h q1  = *(const v8h*)(B1 + k0 + 8);
    v8h p2  = *(const v8h*)(B2 + k0);
    v8h q2  = *(const v8h*)(B2 + k0 + 8);
    v8h p3  = *(const v8h*)(B3 + k0);
    v8h q3  = *(const v8h*)(B3 + k0 + 8);
    Frag f;
    f.a  = cat16(alo, ahi);
    f.b0 = cat16(p0, q0);
    f.b1 = cat16(p1, q1);
    f.b2 = cat16(p2, q2);
    f.b3 = cat16(p3, q3);
    return f;
}

// ---------------------------------------------------------------------------
// WMMA GEMM: D[M,N] (f32) = A[M,K] (f16 row-major) * Bt[N,K] (f16 row-major)
// Both operands K-major -> every fragment is contiguous global b128 loads:
// no LDS staging, no block barriers. 256 threads = 8 wave32 stacked in M;
// each wave owns a 16x64 tile (4 WMMA per 32-K step, A fragment reused 4x).
// Double-buffered software pipeline (K % 64 == 0 for every layer here):
// loads of step k+32 are in flight while step k's WMMAs execute, so WMMAs
// only need partial s_wait_loadcnt values in steady state.
// mode: 0=none, 1=+bias, 2=+bias,relu, 3=+bias,lrelu(0.2)
// ---------------------------------------------------------------------------
__global__ __launch_bounds__(256) void gemm_f16_wmma(
    const _Float16* __restrict__ A, const _Float16* __restrict__ Bt,
    const float* __restrict__ bias, float* __restrict__ D,
    int M, int N, int K, long batchA, long batchB, long batchD, int mode)
{
    const int bz = blockIdx.z;
    A  += (long)bz * batchA;
    Bt += (long)bz * batchB;
    D  += (long)bz * batchD;

    const int tid   = threadIdx.x;
    const int wave  = tid >> 5;                   // 0..7 -> M direction
    const int lane  = tid & 31;
    const int tileM = blockIdx.y * 128 + wave * 16;
    const int tileN = blockIdx.x * 64;

    const int r     = lane & 15;                  // row within A tile / col within B tile
    const int khalf = lane >> 4;                  // ISA K-half selector

    // A fragment: lanes 0-15 K=0..15 contiguous, lanes 16-31 K=16..31
    const _Float16* Arow = A + (long)(tileM + r) * K + khalf * 16;
    // B fragment: lanes 0-15 K=0..15 contiguous, lanes 16-31 K=16..31
    const _Float16* Brow0 = Bt + (long)(tileN +  0 + r) * K + khalf * 16;
    const _Float16* Brow1 = Bt + (long)(tileN + 16 + r) * K + khalf * 16;
    const _Float16* Brow2 = Bt + (long)(tileN + 32 + r) * K + khalf * 16;
    const _Float16* Brow3 = Bt + (long)(tileN + 48 + r) * K + khalf * 16;

    // bias folded into accumulator init (column fixed per lane per tile)
    v8f acc0 = {}, acc1 = {}, acc2 = {}, acc3 = {};
    if (mode >= 1 && bias) {
        float b0 = bias[tileN +  0 + r], b1 = bias[tileN + 16 + r];
        float b2 = bias[tileN + 32 + r], b3 = bias[tileN + 48 + r];
#pragma unroll
        for (int i = 0; i < 8; ++i) { acc0[i] = b0; acc1[i] = b1; acc2[i] = b2; acc3[i] = b3; }
    }

    // ---- software pipeline: F0 holds step k0, F1 holds step k0+32 ----
    Frag f0 = load_frag(Arow, Brow0, Brow1, Brow2, Brow3, 0);

    for (int k0 = 0; k0 < K; k0 += 64) {
        // issue next half-step's loads; they fly during f0's WMMAs
        Frag f1 = load_frag(Arow, Brow0, Brow1, Brow2, Brow3, k0 + 32);
        __builtin_prefetch(Arow  + k0 + 64, 0, 1);
        __builtin_prefetch(Brow0 + k0 + 64, 0, 1);

        acc0 = __builtin_amdgcn_wmma_f32_16x16x32_f16(false, f0.a, false, f0.b0, (short)0, acc0, false, false);
        acc1 = __builtin_amdgcn_wmma_f32_16x16x32_f16(false, f0.a, false, f0.b1, (short)0, acc1, false, false);
        acc2 = __builtin_amdgcn_wmma_f32_16x16x32_f16(false, f0.a, false, f0.b2, (short)0, acc2, false, false);
        acc3 = __builtin_amdgcn_wmma_f32_16x16x32_f16(false, f0.a, false, f0.b3, (short)0, acc3, false, false);

        if (k0 + 64 < K) {
            // refill f0 for the next iteration; flies during f1's WMMAs
            f0 = load_frag(Arow, Brow0, Brow1, Brow2, Brow3, k0 + 64);
        }

        acc0 = __builtin_amdgcn_wmma_f32_16x16x32_f16(false, f1.a, false, f1.b0, (short)0, acc0, false, false);
        acc1 = __builtin_amdgcn_wmma_f32_16x16x32_f16(false, f1.a, false, f1.b1, (short)0, acc1, false, false);
        acc2 = __builtin_amdgcn_wmma_f32_16x16x32_f16(false, f1.a, false, f1.b2, (short)0, acc2, false, false);
        acc3 = __builtin_amdgcn_wmma_f32_16x16x32_f16(false, f1.a, false, f1.b3, (short)0, acc3, false, false);
    }

    // C/D layout: VGPR i -> row (khalf*8 + i), col = lane&15 within each tile
    const int rbase = tileM + khalf * 8;
    float* Dr = D + (long)rbase * N + tileN + r;
#pragma unroll
    for (int i = 0; i < 8; ++i) {
        float v0 = acc0[i], v1 = acc1[i], v2 = acc2[i], v3 = acc3[i];
        if (mode == 2) {
            v0 = fmaxf(v0, 0.0f); v1 = fmaxf(v1, 0.0f);
            v2 = fmaxf(v2, 0.0f); v3 = fmaxf(v3, 0.0f);
        } else if (mode == 3) {
            v0 = (v0 >= 0.0f) ? v0 : 0.2f * v0; v1 = (v1 >= 0.0f) ? v1 : 0.2f * v1;
            v2 = (v2 >= 0.0f) ? v2 : 0.2f * v2; v3 = (v3 >= 0.0f) ? v3 : 0.2f * v3;
        }
        Dr[(long)i * N +  0] = v0;
        Dr[(long)i * N + 16] = v1;
        Dr[(long)i * N + 32] = v2;
        Dr[(long)i * N + 48] = v3;
    }
}

// ---------------------------------------------------------------------------
// im2col: builds A[m, kOffset + kk] in f16, fusing the preceding activation.
// Handles conv (rdil) and transposed conv (ldil = lhs dilation, pad = k-1-p).
// act: 0 none, 1 relu, 2 lrelu(0.2)
// ---------------------------------------------------------------------------
__global__ void im2col_k(
    const float* __restrict__ src, _Float16* __restrict__ A,
    int Cin, int inH, int inW, int outH, int outW,
    int KH, int KW, int stride, int pad, int rdil, int ldil,
    long sB, long sC, long sY, long sX,
    int Ktotal, int kOffset, int act, int total)
{
    int idx = blockIdx.x * blockDim.x + threadIdx.x;
    if (idx >= total) return;
    const int kdim = Cin * KH * KW;
    int kk = idx % kdim;
    int m  = idx / kdim;
    int kx = kk % KW;  int t  = kk / KW;
    int ky = t % KH;   int ci = t / KH;
    int x  = m % outW; int t2 = m / outW;
    int y  = t2 % outH; int bb = t2 / outH;

    int ty = y * stride - pad + ky * rdil;
    int tx = x * stride - pad + kx * rdil;
    float v = 0.0f;
    if (ty >= 0 && tx >= 0 && (ty % ldil) == 0 && (tx % ldil) == 0) {
        int iy = ty / ldil, ix = tx / ldil;
        if (iy < inH && ix < inW)
            v = src[(long)bb * sB + (long)ci * sC + (long)iy * sY + (long)ix * sX];
    }
    if (act == 1)      v = fmaxf(v, 0.0f);
    else if (act == 2) v = (v >= 0.0f) ? v : 0.2f * v;
    A[(long)m * Ktotal + kOffset + kk] = (_Float16)v;
}

// OIHW f32 weights are already [co][k] with k = ci*KH*KW+ky*KW+kx -> pure convert
__global__ void wcvt_k(const float* __restrict__ w, _Float16* __restrict__ Bt, int total)
{
    int idx = blockIdx.x * blockDim.x + threadIdx.x;
    if (idx >= total) return;
    Bt[idx] = (_Float16)w[idx];
}

// ---------------------------------------------------------------------------
// BatchNorm (batch statistics) over channel-last [M][C] activations.
// ---------------------------------------------------------------------------
__global__ __launch_bounds__(256) void bn_stats_k(
    const float* __restrict__ X, float* __restrict__ mean,
    float* __restrict__ rstd, int M, int C)
{
    __shared__ float s1[256], s2[256];
    int c = blockIdx.x, tid = threadIdx.x;
    float a = 0.0f, b = 0.0f;
    for (int m = tid; m < M; m += 256) {
        float v = X[(long)m * C + c];
        a += v; b += v * v;
    }
    s1[tid] = a; s2[tid] = b; __syncthreads();
    for (int s = 128; s > 0; s >>= 1) {
        if (tid < s) { s1[tid] += s1[tid + s]; s2[tid] += s2[tid + s]; }
        __syncthreads();
    }
    if (tid == 0) {
        float mu  = s1[0] / M;
        float var = s2[0] / M - mu * mu;
        mean[c] = mu;
        rstd[c] = rsqrtf(var + EPS_BN);
    }
}

__global__ void bn_apply_k(float* __restrict__ X,
    const float* __restrict__ mean, const float* __restrict__ rstd,
    const float* __restrict__ g, const float* __restrict__ bb,
    int C, int act, int total)
{
    int idx = blockIdx.x * blockDim.x + threadIdx.x;
    if (idx >= total) return;
    int c = idx % C;
    float v = (X[idx] - mean[c]) * rstd[c] * g[c] + bb[c];
    if (act == 1)      v = fmaxf(v, 0.0f);
    else if (act == 2) v = (v >= 0.0f) ? v : 0.2f * v;
    X[idx] = v;
}

// BN5 applied while transposing channel-last -> NCHW into d_out (channels 0..C)
__global__ void bn_out_k(const float* __restrict__ X, float* __restrict__ out,
    const float* __restrict__ mean, const float* __restrict__ rstd,
    const float* __restrict__ g, const float* __restrict__ bb,
    int H, int W, int C, int Ctot, int total)
{
    int idx = blockIdx.x * blockDim.x + threadIdx.x;
    if (idx >= total) return;
    int x = idx % W; int t = idx / W;
    int y = t % H;   t /= H;
    int c = t % C;   int b = t / C;
    float v = X[(((long)b * H + y) * W + x) * C + c];
    v = (v - mean[c]) * rstd[c] * g[c] + bb[c];
    out[(((long)b * Ctot + c) * H + y) * W + x] = v;
}

// append original x (NCHW) at channel offset Coff of d_out
__global__ void concat_x_k(const float* __restrict__ x, float* __restrict__ out,
                           int Cx, int HW, int Ctot, int Coff, int total)
{
    int idx = blockIdx.x * blockDim.x + threadIdx.x;
    if (idx >= total) return;
    int p = idx % HW; int t = idx / HW;
    int c = t % Cx;   int b = t / Cx;
    out[((long)b * Ctot + Coff + c) * HW + p] = x[idx];
}

// ---------------------------------------------------------------------------
// CSA shift pieces. h2 is channel-last [b][HW][C].
// csa_pack: L2-normalize a gathered row and emit f16 [b][n][C] (K-major for
// both GEMM operands, so all writes are contiguous).
// ---------------------------------------------------------------------------
__global__ __launch_bounds__(256) void csa_pack_k(
    const float* __restrict__ h2, const int* __restrict__ idxs,
    _Float16* __restrict__ dst, int C, int HW, int Ncnt)
{
    __shared__ float sred[256];
    __shared__ float rbc;
    int n = blockIdx.x, b = blockIdx.y, tid = threadIdx.x;
    const float* row = h2 + ((long)b * HW + idxs[n]) * C;
    float ss = 0.0f;
    for (int c = tid; c < C; c += 256) { float v = row[c]; ss += v * v; }
    sred[tid] = ss; __syncthreads();
    for (int s = 128; s > 0; s >>= 1) {
        if (tid < s) sred[tid] += sred[tid + s];
        __syncthreads();
    }
    if (tid == 0) rbc = rsqrtf(sred[0] + EPS_CSA);
    __syncthreads();
    float r = rbc;
    _Float16* d = dst + ((long)b * Ncnt + n) * C;
    for (int c = tid; c < C; c += 256) d[c] = (_Float16)(row[c] * r);
}

__global__ __launch_bounds__(256) void csa_argmax_k(
    const float* __restrict__ sim, float* __restrict__ dmax,
    int* __restrict__ jstar, int Nh, int Nk)
{
    __shared__ float sv[256];
    __shared__ int   si[256];
    int n = blockIdx.x, b = blockIdx.y, tid = threadIdx.x;
    const float* srow = sim + ((long)b * Nh + n) * Nk;
    float best = -3.4e38f; int bj = Nk;
    for (int k = tid; k < Nk; k += 256) {
        float v = srow[k];
        if (v > best) { best = v; bj = k; }
    }
    sv[tid] = best; si[tid] = bj; __syncthreads();
    for (int s = 128; s > 0; s >>= 1) {
        if (tid < s) {
            if (sv[tid + s] > sv[tid] ||
                (sv[tid + s] == sv[tid] && si[tid + s] < si[tid])) {
                sv[tid] = sv[tid + s]; si[tid] = si[tid + s];
            }
        }
        __syncthreads();
    }
    if (tid == 0) {
        dmax[(long)b * Nh + n]  = fmaxf(sv[0], 0.0f);
        jstar[(long)b * Nh + n] = si[0];
    }
}

// sequential scan over hole positions; one block per batch; C=512 -> 2 ch/thread
__global__ __launch_bounds__(256) void csa_scan_k(
    float* __restrict__ h2, const int* __restrict__ hole,
    const int* __restrict__ known, const float* __restrict__ dmax,
    const int* __restrict__ jstar, int C, int HW, int Nh)
{
    __shared__ float r0[256], r1[256], r2[256];
    int b = blockIdx.x, tid = threadIdx.x;
    int c0 = tid, c1 = tid + 256;
    float m0 = 0.0f, m1 = 0.0f;
    for (int n = 0; n < Nh; ++n) {
        float* xi = h2 + ((long)b * HW + hole[n]) * C;
        const float* bi = h2 + ((long)b * HW + known[jstar[(long)b * Nh + n]]) * C;
        float x0 = xi[c0], x1 = xi[c1];
        float bb0 = bi[c0], bb1 = bi[c1];
        r0[tid] = m0 * x0 + m1 * x1;
        r1[tid] = m0 * m0 + m1 * m1;
        r2[tid] = x0 * x0 + x1 * x1;
        __syncthreads();
        for (int s = 128; s > 0; s >>= 1) {
            if (tid < s) { r0[tid] += r0[tid+s]; r1[tid] += r1[tid+s]; r2[tid] += r2[tid+s]; }
            __syncthreads();
        }
        float num = r0[0], msq = r1[0], xsq = r2[0];
        float den = sqrtf(fmaxf(msq * xsq, 0.0f)) + EPS_CSA;
        float dad = fmaxf(num / den, 0.0f);
        float dm  = dmax[(long)b * Nh + n];
        float w   = dad + dm + EPS_CSA;
        m0 = (dad * m0 + dm * bb0) / w;
        m1 = (dad * m1 + dm * bb1) / w;
        xi[c0] = m0; xi[c1] = m1;      // in-place: hole n never re-read later
        __syncthreads();
    }
}

// ---------------------------------------------------------------------------
extern "C" void kernel_launch(void* const* d_in, const int* in_sizes, int n_in,
                              void* d_out, int out_size, void* d_ws, size_t ws_size,
                              hipStream_t stream)
{
    (void)in_sizes; (void)n_in; (void)out_size; (void)ws_size;
    const float* x         = (const float*)d_in[0];
    const float* w_down    = (const float*)d_in[1];
    const float* b_down    = (const float*)d_in[2];
    const float* g1        = (const float*)d_in[3];
    const float* b1        = (const float*)d_in[4];
    const float* w_down3   = (const float*)d_in[5];
    const float* b_down3   = (const float*)d_in[6];
    const float* g2        = (const float*)d_in[7];
    const float* b2        = (const float*)d_in[8];
    const float* w_in_down = (const float*)d_in[9];
    const float* b_in_down = (const float*)d_in[10];
    const float* w_in_up   = (const float*)d_in[11];
    const float* b_in_up   = (const float*)d_in[12];
    const float* g3        = (const float*)d_in[13];
    const float* b3        = (const float*)d_in[14];
    const float* w_up3     = (const float*)d_in[15];
    const float* b_up3     = (const float*)d_in[16];
    const float* g4        = (const float*)d_in[17];
    const float* b4        = (const float*)d_in[18];
    const float* w_up      = (const float*)d_in[19];
    const float* b_up      = (const float*)d_in[20];
    const float* g5        = (const float*)d_in[21];
    const float* b5        = (const float*)d_in[22];
    const int*   hole_idx  = (const int*)d_in[23];
    const int*   known_idx = (const int*)d_in[24];
    float* out = (float*)d_out;

    // --- workspace layout -------------------------------------------------
    size_t off = 0;
    auto alloc = [&](size_t bytes) -> void* {
        void* p = (char*)d_ws + off;
        off += (bytes + 255) & ~(size_t)255;
        return p;
    };
    _Float16* Abuf  = (_Float16*)alloc(sizeof(_Float16) * (size_t)8192 * 4096); // max im2col
    _Float16* Bbuf  = (_Float16*)alloc(sizeof(_Float16) * (size_t)512 * 9216);  // max weight mat
    float* h1   = (float*)alloc(sizeof(float) * 2048 * 256);
    float* h2   = (float*)alloc(sizeof(float) * 2048 * 512);
    float* s3   = (float*)alloc(sizeof(float) * 512  * 512);
    float* s4   = (float*)alloc(sizeof(float) * 2048 * 512);
    float* u5   = (float*)alloc(sizeof(float) * 2048 * 256);
    float* u6   = (float*)alloc(sizeof(float) * 8192 * 256);
    _Float16* Asim = (_Float16*)alloc(sizeof(_Float16) * 2 * 256 * 512);
    _Float16* Bsim = (_Float16*)alloc(sizeof(_Float16) * 2 * 768 * 512);
    float* sim   = (float*)alloc(sizeof(float) * 2 * 256 * 768);
    float* dmaxb = (float*)alloc(sizeof(float) * 2 * 256);
    int*   jstarb = (int*)alloc(sizeof(int)   * 2 * 256);
    float* meanb = (float*)alloc(sizeof(float) * 512);
    float* rstdb = (float*)alloc(sizeof(float) * 512);

    auto im2col = [&](const float* src, _Float16* A, int Cin, int inH, int inW,
                      int outH, int outW, int KH, int KW, int stride, int pad,
                      int rdil, int ldil, long sB, long sC, long sY, long sX,
                      int Ktot, int kOff, int act) {
        int total = 2 * outH * outW * Cin * KH * KW;
        im2col_k<<<dim3((total + 255) / 256), dim3(256), 0, stream>>>(
            src, A, Cin, inH, inW, outH, outW, KH, KW, stride, pad, rdil, ldil,
            sB, sC, sY, sX, Ktot, kOff, act, total);
    };
    auto wcvt = [&](const float* w, int elems) {
        wcvt_k<<<dim3((elems + 255) / 256), dim3(256), 0, stream>>>(w, Bbuf, elems);
    };
    auto gemm = [&](const _Float16* A, const _Float16* Bt, const float* bias,
                    float* D, int M, int N, int K, int batch,
                    long bA, long bB, long bD, int mode) {
        gemm_f16_wmma<<<dim3(N / 64, M / 128, batch), dim3(256), 0, stream>>>(
            A, Bt, bias, D, M, N, K, bA, bB, bD, mode);
    };
    auto bn = [&](float* X, const float* g, const float* bb, int M, int C, int act) {
        bn_stats_k<<<dim3(C), dim3(256), 0, stream>>>(X, meanb, rstdb, M, C);
        int total = M * C;
        bn_apply_k<<<dim3((total + 255) / 256), dim3(256), 0, stream>>>(
            X, meanb, rstdb, g, bb, C, act, total);
    };

    // --- L1: h = conv(lrelu(x), w_down, s2, p3, d2): (2,256,64,64)->(2,256,32,32)
    im2col(x, Abuf, 256, 64, 64, 32, 32, 4, 4, /*s*/2, /*p*/3, /*rd*/2, /*ld*/1,
           256L * 4096, 4096, 64, 1, 4096, 0, /*lrelu*/2);
    wcvt(w_down, 256 * 4096);
    gemm(Abuf, Bbuf, b_down, h1, 2048, 256, 4096, 1, 0, 0, 0, 1);
    bn(h1, g1, b1, 2048, 256, /*lrelu fused*/2);

    // --- L2: h = conv3x3(h1) -> (2,512,32,32), channel-last in h2
    im2col(h1, Abuf, 256, 32, 32, 32, 32, 3, 3, 1, 1, 1, 1,
           1024L * 256, 1, 32L * 256, 256, 2304, 0, 0);
    wcvt(w_down3, 512 * 2304);
    gemm(Abuf, Bbuf, b_down3, h2, 2048, 512, 2304, 1, 0, 0, 0, 1);

    // --- CSA shift on h2 (before BN2, as in reference)
    csa_pack_k<<<dim3(256, 2), dim3(256), 0, stream>>>(h2, hole_idx,  Asim, 512, 1024, 256);
    csa_pack_k<<<dim3(768, 2), dim3(256), 0, stream>>>(h2, known_idx, Bsim, 512, 1024, 768);
    gemm(Asim, Bsim, nullptr, sim, 256, 768, 512, 2,
         256L * 512, 768L * 512, 256L * 768, 0);
    csa_argmax_k<<<dim3(256, 2), dim3(256), 0, stream>>>(sim, dmaxb, jstarb, 256, 768);
    csa_scan_k<<<dim3(2), dim3(256), 0, stream>>>(h2, hole_idx, known_idx,
                                                  dmaxb, jstarb, 512, 1024, 256);
    bn(h2, g2, b2, 2048, 512, /*no act: h reused in concat*/0);

    // --- L3: s = relu(conv(lrelu(h2), w_in_down, s2, p3, d2)): 32->16
    im2col(h2, Abuf, 512, 32, 32, 16, 16, 4, 4, 2, 3, 2, 1,
           1024L * 512, 1, 32L * 512, 512, 8192, 0, /*lrelu*/2);
    wcvt(w_in_down, 512 * 8192);
    gemm(Abuf, Bbuf, b_in_down, s3, 512, 512, 8192, 1, 0, 0, 0, /*bias+relu*/2);

    // --- L4: s = conv_t(s3, w_in_up, s2, k4, p1): 16->32 (pad'=2, lhs-dil=2)
    im2col(s3, Abuf, 512, 16, 16, 32, 32, 4, 4, 1, 2, 1, 2,
           256L * 512, 1, 16L * 512, 512, 8192, 0, 0);
    wcvt(w_in_up, 512 * 8192);
    gemm(Abuf, Bbuf, b_in_up, s4, 2048, 512, 8192, 1, 0, 0, 0, 1);
    bn(s4, g3, b3, 2048, 512, 0);

    // --- L5: u = conv_t(relu(concat[s4, h2]), w_up3, s1, k3, p1) -> (2,256,32,32)
    // concat realized as two im2col passes into disjoint K ranges (relu fused)
    im2col(s4, Abuf, 512, 32, 32, 32, 32, 3, 3, 1, 1, 1, 1,
           1024L * 512, 1, 32L * 512, 512, 9216, 0, /*relu*/1);
    im2col(h2, Abuf, 512, 32, 32, 32, 32, 3, 3, 1, 1, 1, 1,
           1024L * 512, 1, 32L * 512, 512, 9216, 4608, /*relu*/1);
    wcvt(w_up3, 256 * 9216);
    gemm(Abuf, Bbuf, b_up3, u5, 2048, 256, 9216, 1, 0, 0, 0, 1);
    bn(u5, g4, b4, 2048, 256, 0);

    // --- L6: u = conv_t(relu(u5), w_up, s2, k4, p1): 32->64 (pad'=2, lhs-dil=2)
    im2col(u5, Abuf, 256, 32, 32, 64, 64, 4, 4, 1, 2, 1, 2,
           1024L * 256, 1, 32L * 256, 256, 4096, 0, /*relu*/1);
    wcvt(w_up, 256 * 4096);
    gemm(Abuf, Bbuf, b_up, u6, 8192, 256, 4096, 1, 0, 0, 0, 1);

    // --- BN5 while transposing to NCHW d_out[:, 0:256], then append x
    bn_stats_k<<<dim3(256), dim3(256), 0, stream>>>(u6, meanb, rstdb, 8192, 256);
    int totO = 2 * 256 * 64 * 64;
    bn_out_k<<<dim3((totO + 255) / 256), dim3(256), 0, stream>>>(
        u6, out, meanb, rstdb, g5, b5, 64, 64, 256, 512, totO);
    concat_x_k<<<dim3((totO + 255) / 256), dim3(256), 0, stream>>>(
        x, out, 256, 4096, 512, 256, totO);
}
